// SparseFactorisationDense_76682346103524
// MI455X (gfx1250) — compile-verified, matrix operand order checked by reading, not run on hardware
//
#include <hip/hip_runtime.h>
#include <stdint.h>

typedef __bf16 v16bf __attribute__((ext_vector_type(16)));
typedef float  v8f   __attribute__((ext_vector_type(8)));

__device__ __forceinline__ unsigned short f32_to_bf16_rne(float f) {
  union { float f; unsigned u; } in; in.f = f;
  unsigned u = in.u;
  u += 0x7FFFu + ((u >> 16) & 1u);   // round-to-nearest-even
  return (unsigned short)(u >> 16);
}

union FragBF16 { uint4 q[2]; v16bf v; };

// ---------------- cast fp32 -> bf16, 8 elems/thread ----------------
__global__ __launch_bounds__(256)
void cast_f32_bf16_kernel(const float* __restrict__ src,
                          unsigned short* __restrict__ dst, int n) {
  int i = (blockIdx.x * 256 + threadIdx.x) * 8;
  if (i + 8 > n) return;
  const float4* s = reinterpret_cast<const float4*>(src + i);
  float4 a = s[0];
  float4 b = s[1];
  unsigned short h[8];
  h[0] = f32_to_bf16_rne(a.x); h[1] = f32_to_bf16_rne(a.y);
  h[2] = f32_to_bf16_rne(a.z); h[3] = f32_to_bf16_rne(a.w);
  h[4] = f32_to_bf16_rne(b.x); h[5] = f32_to_bf16_rne(b.y);
  h[6] = f32_to_bf16_rne(b.z); h[7] = f32_to_bf16_rne(b.w);
  uint4 o;
  o.x = (unsigned)h[0] | ((unsigned)h[1] << 16);
  o.y = (unsigned)h[2] | ((unsigned)h[3] << 16);
  o.z = (unsigned)h[4] | ((unsigned)h[5] << 16);
  o.w = (unsigned)h[6] | ((unsigned)h[7] << 16);
  *reinterpret_cast<uint4*>(dst + i) = o;
}

// ------- fused mask-mul + bf16 convert + transpose: Wt[n][k] = bf16(k[k][n]*m[k][n]) -------
__global__ __launch_bounds__(256)
void prep_weight_kernel(const float* __restrict__ kw,
                        const float* __restrict__ mw,
                        unsigned short* __restrict__ wt,
                        int K, int N) {
  __shared__ unsigned short tile[32][33];
  const int nb = blockIdx.x * 32;
  const int kb = blockIdx.y * 32;
  const int tx = threadIdx.x;        // 0..31
  const int ty = threadIdx.y;        // 0..7
#pragma unroll
  for (int i = 0; i < 32; i += 8) {
    int kk = kb + ty + i;
    int nn = nb + tx;
    float v = kw[(size_t)kk * N + nn] * mw[(size_t)kk * N + nn];
    tile[ty + i][tx] = f32_to_bf16_rne(v);
  }
  __syncthreads();
#pragma unroll
  for (int i = 0; i < 32; i += 8) {
    int nn = nb + ty + i;
    int kk = kb + tx;
    wt[(size_t)nn * K + kk] = tile[tx][ty + i];
  }
}

// ---------------- bf16 WMMA GEMM: C[M,N] = A[M,K] * Bt[N,K]^T ----------------
// Block tile 128x256, 8 waves (2x4), each wave 64x64 (4x4 tiles of 16x16).
// Double-buffered LDS fed by CDNA5 async global->LDS copies (ASYNCcnt),
// compute via v_wmma_f32_16x16x32_bf16.
template<int FINAL>
__global__ __launch_bounds__(256)
void gemm_bf16_wmma_kernel(const unsigned short* __restrict__ A,   // [M,K] bf16
                           const unsigned short* __restrict__ Bt,  // [N,K] bf16
                           void* __restrict__ Cout,
                           int M, int N, int K,
                           const float* __restrict__ scaling,
                           const float* __restrict__ bias) {
  constexpr int BM = 128, BN = 256, BK = 32, LDA = BK + 8;  // 40 elems = 80B row
  constexpr unsigned ABUF = (unsigned)(BM * LDA * 2);       // bytes per A buffer
  constexpr unsigned BBUF = (unsigned)(BN * LDA * 2);       // bytes per B buffer
  __shared__ __align__(16) unsigned short As[2][BM * LDA];
  __shared__ __align__(16) unsigned short Bs[2][BN * LDA];

  const int tid   = threadIdx.x;
  const int lane  = tid & 31;
  const int wave  = tid >> 5;        // 0..7
  const int wm    = wave >> 2;       // 0..1 -> 64-row slab
  const int wn    = wave & 3;        // 0..3 -> 64-col slab
  const int l16   = lane & 15;
  const int khalf = lane >> 4;       // 0..1
  const int mblk  = blockIdx.y * BM;
  const int nblk  = blockIdx.x * BN;

  // Per-thread chunk assignment: A = 512 x 16B chunks (2/thread), B = 1024 (4/thread).
  // Precompute global byte offsets (without K step) and LDS byte addresses (buf 0).
  const unsigned asBase = (unsigned)(uintptr_t)(&As[0][0]);  // LDS aperture: low 32 bits
  const unsigned bsBase = (unsigned)(uintptr_t)(&Bs[0][0]);
  unsigned a_goff[2], a_lds[2], b_goff[4], b_lds[4];
#pragma unroll
  for (int j = 0; j < 2; ++j) {
    int c = tid + 256 * j;
    int r = c >> 2, c8 = (c & 3) * 8;
    a_goff[j] = (unsigned)(((size_t)(mblk + r) * K + c8) * 2);
    a_lds[j]  = asBase + (unsigned)((r * LDA + c8) * 2);
  }
#pragma unroll
  for (int j = 0; j < 4; ++j) {
    int c = tid + 256 * j;
    int r = c >> 2, c8 = (c & 3) * 8;
    b_goff[j] = (unsigned)(((size_t)(nblk + r) * K + c8) * 2);
    b_lds[j]  = bsBase + (unsigned)((r * LDA + c8) * 2);
  }

  auto issueAsync = [&](int kt, int buf) {
    const unsigned kb2 = (unsigned)(kt * BK * 2);   // byte step along K
    const unsigned ao  = (unsigned)buf * ABUF;
    const unsigned bo  = (unsigned)buf * BBUF;
#pragma unroll
    for (int j = 0; j < 2; ++j) {
      unsigned voff = a_goff[j] + kb2;
      unsigned lds  = a_lds[j] + ao;
      asm volatile("global_load_async_to_lds_b128 %0, %1, %2"
                   :: "v"(lds), "v"(voff), "s"(A) : "memory");
    }
#pragma unroll
    for (int j = 0; j < 4; ++j) {
      unsigned voff = b_goff[j] + kb2;
      unsigned lds  = b_lds[j] + bo;
      asm volatile("global_load_async_to_lds_b128 %0, %1, %2"
                   :: "v"(lds), "v"(voff), "s"(Bt) : "memory");
    }
  };

  v8f acc[4][4];
  const v8f vzero = {0.f, 0.f, 0.f, 0.f, 0.f, 0.f, 0.f, 0.f};
#pragma unroll
  for (int i = 0; i < 4; ++i)
#pragma unroll
    for (int j = 0; j < 4; ++j) acc[i][j] = vzero;

  const int nk = K / BK;
  issueAsync(0, 0);

  for (int kt = 0; kt < nk; ++kt) {
    const int buf = kt & 1;
    if (kt + 1 < nk) {
      issueAsync(kt + 1, buf ^ 1);
      // In-order async completion: <=6 outstanding means tile kt has landed.
      asm volatile("s_wait_asynccnt 0x6" ::: "memory");
    } else {
      asm volatile("s_wait_asynccnt 0x0" ::: "memory");
    }
    __syncthreads();   // all waves' tile-kt data visible in LDS

    FragBF16 afrag[4], bfrag[4];
#pragma unroll
    for (int mt = 0; mt < 4; ++mt) {
      int r = wm * 64 + mt * 16 + l16;
      const unsigned short* p = &As[buf][r * LDA + khalf * 8];
      afrag[mt].q[0] = *reinterpret_cast<const uint4*>(p);       // K 0..7 / 8..15
      afrag[mt].q[1] = *reinterpret_cast<const uint4*>(p + 16);  // K 16..23 / 24..31
    }
#pragma unroll
    for (int nt = 0; nt < 4; ++nt) {
      int r = wn * 64 + nt * 16 + l16;
      const unsigned short* p = &Bs[buf][r * LDA + khalf * 8];
      bfrag[nt].q[0] = *reinterpret_cast<const uint4*>(p);
      bfrag[nt].q[1] = *reinterpret_cast<const uint4*>(p + 16);
    }

#pragma unroll
    for (int mt = 0; mt < 4; ++mt)
#pragma unroll
      for (int nt = 0; nt < 4; ++nt)
        acc[mt][nt] = __builtin_amdgcn_wmma_f32_16x16x32_bf16(
            false, afrag[mt].v, false, bfrag[nt].v,
            (short)0, acc[mt][nt], false, false);

    __syncthreads();   // write-after-read: next issue targets buf we just read
  }

  // ---- epilogue ----
  const float sc = FINAL ? scaling[0] : 0.f;
#pragma unroll
  for (int mt = 0; mt < 4; ++mt) {
#pragma unroll
    for (int nt = 0; nt < 4; ++nt) {
      const int col     = nblk + wn * 64 + nt * 16 + l16;
      const int rowbase = mblk + wm * 64 + mt * 16 + khalf * 8;
      const float b = FINAL ? bias[col] : 0.f;
#pragma unroll
      for (int r = 0; r < 8; ++r) {
        const int row = rowbase + r;
        float v = acc[mt][nt][r];
        if (FINAL) {
          v = sc * v + b;
          v = v > 0.f ? v : 0.f;
          reinterpret_cast<float*>(Cout)[(size_t)row * N + col] = v;
        } else {
          reinterpret_cast<unsigned short*>(Cout)[(size_t)row * N + col] =
              f32_to_bf16_rne(v);
        }
      }
    }
  }
}

extern "C" void kernel_launch(void* const* d_in, const int* in_sizes, int n_in,
                              void* d_out, int out_size, void* d_ws, size_t ws_size,
                              hipStream_t stream) {
  const int Bm = 8192, D = 2048, U = 2048;
  const float* x       = (const float*)d_in[0];
  const float* k0      = (const float*)d_in[1];
  const float* k1      = (const float*)d_in[2];
  const float* k2      = (const float*)d_in[3];
  const float* m0      = (const float*)d_in[4];
  const float* m1      = (const float*)d_in[5];
  const float* m2      = (const float*)d_in[6];
  const float* scaling = (const float*)d_in[7];
  const float* bias    = (const float*)d_in[8];

  // Workspace layout (bytes): xb 32M | w0t 8M | w1t 8M | w2t 8M | h0 32M | h1 32M = 120 MB
  char* ws = (char*)d_ws;
  unsigned short* xb  = (unsigned short*)ws; ws += (size_t)Bm * D * 2;
  unsigned short* w0t = (unsigned short*)ws; ws += (size_t)D  * U * 2;
  unsigned short* w1t = (unsigned short*)ws; ws += (size_t)U  * U * 2;
  unsigned short* w2t = (unsigned short*)ws; ws += (size_t)U  * U * 2;
  unsigned short* h0  = (unsigned short*)ws; ws += (size_t)Bm * U * 2;
  unsigned short* h1  = (unsigned short*)ws;

  // 1) cast activations to bf16
  cast_f32_bf16_kernel<<<dim3((Bm * D) / (256 * 8)), dim3(256), 0, stream>>>(x, xb, Bm * D);

  // 2) fold masks into weights, convert to bf16, transpose to [N,K]
  dim3 tblk(32, 8);
  prep_weight_kernel<<<dim3(U / 32, D / 32), tblk, 0, stream>>>(k0, m0, w0t, D, U);
  prep_weight_kernel<<<dim3(U / 32, U / 32), tblk, 0, stream>>>(k1, m1, w1t, U, U);
  prep_weight_kernel<<<dim3(U / 32, U / 32), tblk, 0, stream>>>(k2, m2, w2t, U, U);

  // 3) three WMMA GEMMs (last one fuses scale + bias + relu, writes fp32)
  dim3 ggrid(U / 256, Bm / 128), gblk(256);
  gemm_bf16_wmma_kernel<0><<<ggrid, gblk, 0, stream>>>(xb, w0t, h0, Bm, U, D, scaling, bias);
  gemm_bf16_wmma_kernel<0><<<ggrid, gblk, 0, stream>>>(h0, w1t, h1, Bm, U, U, scaling, bias);
  gemm_bf16_wmma_kernel<1><<<ggrid, gblk, 0, stream>>>(h1, w2t, d_out, Bm, U, U, scaling, bias);
}
